// MultiHeadSelfAttention_41171556500240
// MI455X (gfx1250) — matrix operand order repaired and checked by reading, not compiled, not run on hardware
//
#include <hip/hip_runtime.h>
#include <hip/hip_bf16.h>

typedef unsigned short u16;
typedef unsigned int   u32;
typedef __bf16 v16bf __attribute__((ext_vector_type(16)));
typedef float  v8f   __attribute__((ext_vector_type(8)));
typedef u32    v4u   __attribute__((ext_vector_type(4)));

union Frag16 { v16bf bf; v4u q[2]; };

// ---------------- helpers ----------------

static __device__ __forceinline__ u16 cvt_bf16(float f) {
  u32 u = __float_as_uint(f);
  u32 r = (u + 0x7FFFu + ((u >> 16) & 1u)) >> 16;   // round-to-nearest-even
  return (u16)r;
}

static __device__ __forceinline__ v8f wmma_bf16(const Frag16& a, const Frag16& b, v8f c) {
  return __builtin_amdgcn_wmma_f32_16x16x32_bf16(false, a.bf, false, b.bf, (short)0, c,
                                                 false, false);
}

// CDNA5 async copy: global -> LDS, 16B per lane, tracked by ASYNCcnt.
static __device__ __forceinline__ void async_ld_b128(const u16* g, u16* lds_generic) {
  u32 lds_off = (u32)(unsigned long long)(uintptr_t)lds_generic;  // low 32 bits = LDS offset
  asm volatile("global_load_async_to_lds_b128 %0, %1, off"
               :: "v"(lds_off), "v"(g) : "memory");
}
static __device__ __forceinline__ void wait_async0() {
  asm volatile("s_wait_asynccnt 0x0" ::: "memory");
}

static __device__ __forceinline__ float redmax16(float v) {
  v = fmaxf(v, __shfl_xor(v, 1, 32));
  v = fmaxf(v, __shfl_xor(v, 2, 32));
  v = fmaxf(v, __shfl_xor(v, 4, 32));
  v = fmaxf(v, __shfl_xor(v, 8, 32));
  return v;
}
static __device__ __forceinline__ float redsum16(float v) {
  v += __shfl_xor(v, 1, 32);
  v += __shfl_xor(v, 2, 32);
  v += __shfl_xor(v, 4, 32);
  v += __shfl_xor(v, 8, 32);
  return v;
}

// ---------------- fp32 -> bf16 convert / transpose ----------------

__global__ void cvt_bf16_kernel(const float* __restrict__ in, u16* __restrict__ out, long n) {
  long i = (long)blockIdx.x * blockDim.x + threadIdx.x;
  if (i < n) out[i] = cvt_bf16(in[i]);
}

// WT[n*K + k] = bf16(W[k*N + n])
__global__ void transpose_cvt_kernel(const float* __restrict__ W, u16* __restrict__ WT,
                                     int K, int N) {
  long idx = (long)blockIdx.x * blockDim.x + threadIdx.x;
  if (idx >= (long)K * N) return;
  int k = (int)(idx / N);
  int n = (int)(idx % N);
  WT[(long)n * K + k] = cvt_bf16(W[idx]);
}

// ---------------- bf16 WMMA GEMM: out[M,N] = A[M,K] @ BT[N,K]^T (+bias) ------
// OUT_MODE 0: bf16 output, no bias.  OUT_MODE 1: f32 output + bias.
// Double-buffered LDS, async-to-LDS pipeline.

template <int OUT_MODE>
__global__ __launch_bounds__(256) void gemm_bf16_kernel(
    const u16* __restrict__ A, const u16* __restrict__ BT,
    const float* __restrict__ bias, void* __restrict__ outp,
    int M, int N, int K) {
  __shared__ __align__(16) u16 ldsA[2][128 * 32];
  __shared__ __align__(16) u16 ldsB[2][128 * 32];

  const int tid  = threadIdx.x;
  const int wave = tid >> 5, lane = tid & 31;
  const int half = lane >> 4, lx = lane & 15;
  const int wm = wave & 3, wn = wave >> 2;       // 4x2 wave grid -> 128x128 tile
  const int m0 = blockIdx.y * 128, n0 = blockIdx.x * 128;

  v8f acc[2][4];
#pragma unroll
  for (int s = 0; s < 2; s++)
#pragma unroll
    for (int f = 0; f < 4; f++)
#pragma unroll
      for (int i = 0; i < 8; i++) acc[s][f][i] = 0.f;

  // per-thread tile slice: 2 x b128 per tile (128x32 bf16 = 512 x b128 / 256 thr)
  const int idx0 = tid * 2;
  const int row0 = idx0 >> 2, c40 = (idx0 & 3) * 8;
  const int idx1 = tid * 2 + 1;
  const int row1 = idx1 >> 2, c41 = (idx1 & 3) * 8;

  // prologue: fill buffer 0
  async_ld_b128(A + (long)(m0 + row0) * K + c40, &ldsA[0][row0 * 32 + c40]);
  async_ld_b128(A + (long)(m0 + row1) * K + c41, &ldsA[0][row1 * 32 + c41]);
  async_ld_b128(BT + (long)(n0 + row0) * K + c40, &ldsB[0][row0 * 32 + c40]);
  async_ld_b128(BT + (long)(n0 + row1) * K + c41, &ldsB[0][row1 * 32 + c41]);
  wait_async0();
  __syncthreads();

  for (int k0 = 0; k0 < K; k0 += 32) {
    const int cur = (k0 >> 5) & 1, nxt = cur ^ 1;
    if (k0 + 32 < K) {
      const int kn = k0 + 32;
      async_ld_b128(A + (long)(m0 + row0) * K + kn + c40, &ldsA[nxt][row0 * 32 + c40]);
      async_ld_b128(A + (long)(m0 + row1) * K + kn + c41, &ldsA[nxt][row1 * 32 + c41]);
      async_ld_b128(BT + (long)(n0 + row0) * K + kn + c40, &ldsB[nxt][row0 * 32 + c40]);
      async_ld_b128(BT + (long)(n0 + row1) * K + kn + c41, &ldsB[nxt][row1 * 32 + c41]);
    }

    Frag16 af[2], bfr[4];
#pragma unroll
    for (int s = 0; s < 2; s++) {
      const u16* p = &ldsA[cur][(wm * 32 + s * 16 + lx) * 32 + half * 8];
      af[s].q[0] = *(const v4u*)p;
      af[s].q[1] = *(const v4u*)(p + 16);
    }
#pragma unroll
    for (int f = 0; f < 4; f++) {
      const u16* p = &ldsB[cur][(wn * 64 + f * 16 + lx) * 32 + half * 8];
      bfr[f].q[0] = *(const v4u*)p;
      bfr[f].q[1] = *(const v4u*)(p + 16);
    }
#pragma unroll
    for (int s = 0; s < 2; s++)
#pragma unroll
      for (int f = 0; f < 4; f++)
        acc[s][f] = wmma_bf16(af[s], bfr[f], acc[s][f]);

    wait_async0();        // next tile fully landed in LDS
    __syncthreads();      // and everyone is done reading 'cur'
  }

#pragma unroll
  for (int s = 0; s < 2; s++)
#pragma unroll
    for (int f = 0; f < 4; f++) {
      int col = n0 + wn * 64 + f * 16 + lx;
      float bv = (OUT_MODE == 1) ? bias[col] : 0.f;
#pragma unroll
      for (int i = 0; i < 8; i++) {
        int row = m0 + wm * 32 + s * 16 + half * 8 + i;
        float v = acc[s][f][i];
        if (OUT_MODE == 1)
          ((float*)outp)[(long)row * N + col] = v + bv;
        else
          ((u16*)outp)[(long)row * N + col] = cvt_bf16(v);
      }
    }
}

// ---------------- flash attention -----------------------------------------
// qkv: bf16 [B*2048][3072], q at col h*64, k at 1024+h*64, v at 2048+h*64
// ctx: bf16 [B*2048][1024], col h*64+d.  grid = (N/64, H, B), 128 threads.

__global__ __launch_bounds__(128) void attn_kernel(const u16* __restrict__ qkv,
                                                   u16* __restrict__ ctx) {
  const int qb = blockIdx.x, h = blockIdx.y, b = blockIdx.z;
  const int tid = threadIdx.x;
  const int wave = tid >> 5, lane = tid & 31;
  const int half = lane >> 4, lx = lane & 15;

  __shared__ __align__(16) u16 ldsK[32 * 64];    // [kv][d]
  __shared__ __align__(16) u16 ldsVT[64 * 32];   // [d][kv]
  __shared__ __align__(16) u16 ldsP[4][16 * 32]; // per-wave P transpose buffer

  const u16* base = qkv + (long)b * 2048 * 3072 + h * 64;
  const int r0 = qb * 64 + wave * 16;            // this wave's first q row

  // Q fragments: 16 rows x 64 d, two k=32 chunks, kept in registers
  Frag16 qf[2];
  {
    const u16* qrow = base + (long)(r0 + lx) * 3072;
#pragma unroll
    for (int c = 0; c < 2; c++) {
      int kb = c * 32 + half * 8;
      qf[c].q[0] = *(const v4u*)(qrow + kb);
      qf[c].q[1] = *(const v4u*)(qrow + kb + 16);
    }
  }

  v8f o[4];
  float m8[8], l8[8];
#pragma unroll
  for (int f = 0; f < 4; f++)
#pragma unroll
    for (int i = 0; i < 8; i++) o[f][i] = 0.f;
#pragma unroll
  for (int i = 0; i < 8; i++) { m8[i] = -3.0e38f; l8[i] = 0.f; }

  const int jmax = qb * 64 + 64;
  for (int j0 = 0; j0 < jmax; j0 += 32) {
    __syncthreads();
    // K tile 32x64 via async DMA to LDS: 256 x b128, 2 per thread
#pragma unroll
    for (int i = 0; i < 2; i++) {
      int idx = tid * 2 + i;
      int row = idx >> 3, c4 = (idx & 7) * 8;
      async_ld_b128(base + 1024 + (long)(j0 + row) * 3072 + c4,
                    &ldsK[row * 64 + c4]);
    }
    // V tile 32x64 transposed into VT[d][kv]: 1024 dwords, 8 per thread
#pragma unroll
    for (int i = 0; i < 8; i++) {
      int idx = tid * 8 + i;
      int row = idx >> 5, cp = idx & 31;
      u32 u = *(const u32*)(base + 2048 + (long)(j0 + row) * 3072 + cp * 2);
      ldsVT[(cp * 2) * 32 + row]     = (u16)(u & 0xffffu);
      ldsVT[(cp * 2 + 1) * 32 + row] = (u16)(u >> 16);
    }
    wait_async0();
    __syncthreads();

    const bool active = (j0 <= r0 + 15);
    if (active) {
      // S = Q @ K^T, two 16x16 frags covering 32 kv cols
      v8f s[2];
#pragma unroll
      for (int f = 0; f < 2; f++) {
        v8f a;
#pragma unroll
        for (int i = 0; i < 8; i++) a[i] = 0.f;
#pragma unroll
        for (int c = 0; c < 2; c++) {
          Frag16 kf;
          const u16* rp = ldsK + (f * 16 + lx) * 64 + c * 32 + half * 8;
          kf.q[0] = *(const v4u*)rp;
          kf.q[1] = *(const v4u*)(rp + 16);
          a = wmma_bf16(qf[c], kf, a);
        }
        s[f] = a;
      }

      // scale + causal mask + online softmax stats (rows = (half*8+i))
      float alpha[8];
      v8f p0, p1;
#pragma unroll
      for (int i = 0; i < 8; i++) {
        int row = r0 + half * 8 + i;
        int c0 = j0 + lx, c1 = j0 + 16 + lx;
        float s0 = s[0][i] * 0.125f + ((c0 > row) ? -1e9f : 0.f);
        float s1 = s[1][i] * 0.125f + ((c1 > row) ? -1e9f : 0.f);
        float t = redmax16(fmaxf(s0, s1));
        float mn = fmaxf(m8[i], t);
        alpha[i] = __expf(m8[i] - mn);
        p0[i] = __expf(s0 - mn);
        p1[i] = __expf(s1 - mn);
        float rs = redsum16(p0[i] + p1[i]);
        l8[i] = l8[i] * alpha[i] + rs;
        m8[i] = mn;
      }
#pragma unroll
      for (int f = 0; f < 4; f++)
#pragma unroll
        for (int i = 0; i < 8; i++) o[f][i] *= alpha[i];

      // transpose P (C-layout -> A-layout) through per-wave LDS
      u16* pp = &ldsP[wave][0];
#pragma unroll
      for (int i = 0; i < 8; i++) {
        pp[(half * 8 + i) * 32 + lx]      = cvt_bf16(p0[i]);
        pp[(half * 8 + i) * 32 + 16 + lx] = cvt_bf16(p1[i]);
      }
      asm volatile("s_wait_dscnt 0" ::: "memory");
      Frag16 pf;
      {
        const u16* pr = pp + lx * 32 + half * 8;
        pf.q[0] = *(const v4u*)pr;
        pf.q[1] = *(const v4u*)(pr + 16);
      }
      // O += P @ V
#pragma unroll
      for (int f = 0; f < 4; f++) {
        Frag16 vf;
        const u16* vp = ldsVT + (f * 16 + lx) * 32 + half * 8;
        vf.q[0] = *(const v4u*)vp;
        vf.q[1] = *(const v4u*)(vp + 16);
        o[f] = wmma_bf16(pf, vf, o[f]);
      }
    }
  }

  float inv[8];
#pragma unroll
  for (int i = 0; i < 8; i++) inv[i] = 1.f / l8[i];
#pragma unroll
  for (int f = 0; f < 4; f++)
#pragma unroll
    for (int i = 0; i < 8; i++) {
      int row = r0 + half * 8 + i;
      ctx[((long)b * 2048 + row) * 1024 + h * 64 + f * 16 + lx] =
          cvt_bf16(o[f][i] * inv[i]);
    }
}

// ---------------- launch ----------------

extern "C" void kernel_launch(void* const* d_in, const int* in_sizes, int n_in,
                              void* d_out, int out_size, void* d_ws, size_t ws_size,
                              hipStream_t stream) {
  const float* x    = (const float*)d_in[0];
  // d_in[1] = attn_mask (causal, applied analytically in-kernel)
  const float* Wqkv = (const float*)d_in[2];
  const float* Wout = (const float*)d_in[3];
  const float* bout = (const float*)d_in[4];
  float* out = (float*)d_out;

  char* ws = (char*)d_ws;
  u16* xb    = (u16*)(ws);                                        // 16 MB
  u16* WqkvT = (u16*)(ws + (size_t)16  * 1024 * 1024);            //  6 MB
  u16* WoutT = (u16*)(ws + (size_t)22  * 1024 * 1024);            //  2 MB
  u16* qkvb  = (u16*)(ws + (size_t)24  * 1024 * 1024);            // 48 MB
  u16* ctxb  = (u16*)(ws + (size_t)72  * 1024 * 1024);            // 16 MB

  // 1) converts / transposes
  cvt_bf16_kernel<<<dim3((8192L * 1024 + 255) / 256), 256, 0, stream>>>(x, xb, 8192L * 1024);
  transpose_cvt_kernel<<<dim3((1024L * 3072 + 255) / 256), 256, 0, stream>>>(Wqkv, WqkvT, 1024, 3072);
  transpose_cvt_kernel<<<dim3((1024L * 1024 + 255) / 256), 256, 0, stream>>>(Wout, WoutT, 1024, 1024);

  // 2) QKV projection: (8192x1024) @ (1024x3072) -> bf16 qkv
  gemm_bf16_kernel<0><<<dim3(3072 / 128, 8192 / 128), 256, 0, stream>>>(
      xb, WqkvT, nullptr, qkvb, 8192, 3072, 1024);

  // 3) flash attention -> bf16 ctx (B,N,C layout)
  attn_kernel<<<dim3(2048 / 64, 16, 4), 128, 0, stream>>>(qkvb, ctxb);

  // 4) output projection + bias -> f32 out
  gemm_bf16_kernel<1><<<dim3(1024 / 128, 8192 / 128), 256, 0, stream>>>(
      ctxb, WoutT, bout, out, 8192, 1024, 1024);
}
